// MultiHeadSelfAttentionDense_45947560133410
// MI455X (gfx1250) — matrix operand order repaired and checked by reading, not compiled, not run on hardware
//
#include <hip/hip_runtime.h>

// ---------------------------------------------------------------------------
// MHA for MI455X (gfx1250, wave32, WMMA 16x16x32 bf16, fp32 accumulate,
// TDM async tensor loads for K/V tile staging with double buffering).
// B=2, S=4096, D=512, H=8, Dh=64.  Compute-bound (~86 GFLOP vs ~44MB HBM);
// per-head K/V (512KB each) stays L2-resident across the 64 query tiles.
// ---------------------------------------------------------------------------

typedef __attribute__((ext_vector_type(16))) __bf16 v16bf;
typedef __attribute__((ext_vector_type(8)))  float  v8f;
typedef __attribute__((ext_vector_type(4)))  unsigned int u32x4;
typedef __attribute__((ext_vector_type(8)))  int          i32x8;
typedef __attribute__((ext_vector_type(4)))  int          i32x4;

union Frag {
    v16bf v;
    uint4 u[2];
};

__device__ __forceinline__ unsigned short f2bf(float f) {
    unsigned int u = __float_as_uint(f);
    unsigned int r = (u + 0x7FFFu + ((u >> 16) & 1u)) >> 16;   // RNE
    return (unsigned short)r;
}

// ---------------------------------------------------------------------------
// TDM: 2D tile load Global -> LDS via Tensor Data Mover (D# per ISA 8.3-8.5).
// data_size = 8 bytes (code 3); all *_units arguments are in 8-byte units.
// Must be executed by a single wave (TDM ignores EXEC).
// ---------------------------------------------------------------------------
__device__ __forceinline__ void tdm_load_2d(unsigned lds_addr,
                                            unsigned long long gaddr,
                                            unsigned tile_d0_units,
                                            unsigned tile_d1,
                                            unsigned tensor_d0_units,
                                            unsigned tensor_d1,
                                            unsigned long long stride0_units)
{
    u32x4 g0;
    g0[0] = 1u;                                          // count=1 (valid), user mode
    g0[1] = lds_addr;                                    // LDS byte address
    g0[2] = (unsigned)(gaddr & 0xFFFFFFFFull);           // global_addr[31:0]
    g0[3] = (unsigned)((gaddr >> 32) & 0x1FFFFFFull)     // global_addr[56:32]
            | (2u << 30);                                // type=2 ("image")

    i32x8 g1;
    g1[0] = (int)(3u << 16);                             // wg_mask=0, data_size=8B
    g1[1] = (int)((tensor_d0_units & 0xFFFFu) << 16);    // tensor_dim0[15:0]
    g1[2] = (int)((tensor_d0_units >> 16) |
                  ((tensor_d1 & 0xFFFFu) << 16));        // tdim0[31:16] | tdim1[15:0]
    g1[3] = (int)((tensor_d1 >> 16) |
                  ((tile_d0_units & 0xFFFFu) << 16));    // tdim1[31:16] | tile_dim0
    g1[4] = (int)(tile_d1 & 0xFFFFu);                    // tile_dim1, tile_dim2=0
    g1[5] = (int)(stride0_units & 0xFFFFFFFFull);        // dim0_stride[31:0]
    g1[6] = (int)((stride0_units >> 32) & 0xFFFFull);    // dim0_stride[47:32]
    g1[7] = 0;

    i32x4 z4;
    z4[0] = 0; z4[1] = 0; z4[2] = 0; z4[3] = 0;
#if defined(__clang_major__) && (__clang_major__ >= 23)
    i32x8 z8;
    z8[0] = 0; z8[1] = 0; z8[2] = 0; z8[3] = 0;
    z8[4] = 0; z8[5] = 0; z8[6] = 0; z8[7] = 0;
    __builtin_amdgcn_tensor_load_to_lds(g0, g1, z4, z4, z8, 0);
#else
    __builtin_amdgcn_tensor_load_to_lds(g0, g1, z4, z4, 0);
#endif
}

// ---------------------------------------------------------------------------
// GEMM: C[8192 x 512] = A[8192 x 512] * W[512 x 512]
//   ABF16: A is bf16 workspace vs fp32 input (converted on the fly)
//   MODE:  0 = bf16 head-split [B,H,S,64] * scale   (Q, K)
//          1 = fp32 plain [M,512]                   (final output)
//          2 = bf16 V-transposed [B,H,64,S]         (V for attention)
// Block 128 threads (4 waves); tile 64x64, K-step 32.
// ---------------------------------------------------------------------------
template<bool ABF16, int MODE>
__global__ __launch_bounds__(128) void gemm512(const void* __restrict__ Aptr,
                                               const float* __restrict__ W,
                                               void* __restrict__ Outp,
                                               float scale)
{
    __shared__ unsigned short As[64][32];    // bf16 A tile, row-major
    __shared__ unsigned short BsT[64][32];   // bf16 W tile, TRANSPOSED [n][k]

    const int t    = threadIdx.x;
    const int lane = t & 31;
    const int wave = t >> 5;
    const int lr   = lane & 15;
    const int lh   = lane >> 4;
    const int m0   = blockIdx.y * 64;
    const int n0   = blockIdx.x * 64;

    v8f acc[4] = {};

    for (int kt = 0; kt < 16; ++kt) {
        const int k0 = kt * 32;
        __syncthreads();
        if (ABF16) {
            const unsigned short* A = (const unsigned short*)Aptr;
            #pragma unroll
            for (int i = 0; i < 2; ++i) {
                int idx = (i * 128 + t) * 8;
                int r = idx >> 5, c = idx & 31;
                *(uint4*)&As[r][c] =
                    *(const uint4*)&A[(size_t)(m0 + r) * 512 + k0 + c];
            }
        } else {
            const float* A = (const float*)Aptr;
            #pragma unroll
            for (int i = 0; i < 4; ++i) {
                int idx = (i * 128 + t) * 4;
                int r = idx >> 5, c = idx & 31;
                const float4 f = *(const float4*)&A[(size_t)(m0 + r) * 512 + k0 + c];
                As[r][c + 0] = f2bf(f.x);
                As[r][c + 1] = f2bf(f.y);
                As[r][c + 2] = f2bf(f.z);
                As[r][c + 3] = f2bf(f.w);
            }
        }
        #pragma unroll
        for (int i = 0; i < 4; ++i) {
            int idx = (i * 128 + t) * 4;
            int r = idx >> 6, c = idx & 63;
            const float4 f = *(const float4*)&W[(size_t)(k0 + r) * 512 + n0 + c];
            BsT[c + 0][r] = f2bf(f.x);
            BsT[c + 1][r] = f2bf(f.y);
            BsT[c + 2][r] = f2bf(f.z);
            BsT[c + 3][r] = f2bf(f.w);
        }
        __syncthreads();

        Frag a;   // row = wave*16+lr; per-lane K = {lh*8..+7, 16+lh*8..+7}
        a.u[0] = *(const uint4*)&As[wave * 16 + lr][lh * 8];
        a.u[1] = *(const uint4*)&As[wave * 16 + lr][16 + lh * 8];
        #pragma unroll
        for (int f = 0; f < 4; ++f) {
            Frag b;   // col = f*16+lr; per-lane K = {lh*8..+7, 16+lh*8..+7}
            b.u[0] = *(const uint4*)&BsT[f * 16 + lr][lh * 8];
            b.u[1] = *(const uint4*)&BsT[f * 16 + lr][16 + lh * 8];
            acc[f] = __builtin_amdgcn_wmma_f32_16x16x32_bf16(
                false, a.v, false, b.v, (short)0, acc[f], false, false);
        }
    }

    // ---- epilogue: C element (j,lane) -> row j + lh*8, col lr ----
    #pragma unroll
    for (int f = 0; f < 4; ++f) {
        if (MODE == 2) {
            // V-transposed: d fixed per (f,lane); s contiguous over j -> b128
            unsigned short tmp[8];
            #pragma unroll
            for (int j = 0; j < 8; ++j) tmp[j] = f2bf(acc[f][j] * scale);
            int mg0 = m0 + wave * 16 + lh * 8;
            int b = mg0 >> 12, s = mg0 & 4095;
            int ng = n0 + f * 16 + lr;
            int h = ng >> 6, d = ng & 63;
            size_t idx = (((size_t)(b * 8 + h) * 64 + d) * 4096) + s;
            *(uint4*)&((unsigned short*)Outp)[idx] = *(const uint4*)tmp;
        } else {
            #pragma unroll
            for (int j = 0; j < 8; ++j) {
                int mg = m0 + wave * 16 + lh * 8 + j;
                int ng = n0 + f * 16 + lr;
                float val = acc[f][j] * scale;
                if (MODE == 1) {
                    ((float*)Outp)[(size_t)mg * 512 + ng] = val;
                } else {  // MODE == 0
                    int b = mg >> 12, s = mg & 4095;
                    int h = ng >> 6,  d = ng & 63;
                    ((unsigned short*)Outp)
                        [(((size_t)(b * 8 + h) * 4096) + s) * 64 + d] = f2bf(val);
                }
            }
        }
    }
}

// ---------------------------------------------------------------------------
// Flash attention: grid (S/64, B*H), 128 threads (4 waves, 16 queries each).
// Q,K bf16 [B,H,S,64] (Q pre-scaled by 1/8); V bf16 TRANSPOSED [B,H,64,S].
// K/V tiles staged by the Tensor Data Mover, double-buffered.
// ---------------------------------------------------------------------------
__global__ __launch_bounds__(128) void attn_kernel(const unsigned short* __restrict__ Q,
                                                   const unsigned short* __restrict__ Kg,
                                                   const unsigned short* __restrict__ Vg,
                                                   unsigned short* __restrict__ Ctx)
{
    __shared__ unsigned short Ks[2][64][64];    // K tile [key][dh], 2 buffers
    __shared__ unsigned short Vs[2][64][64];    // V tile [dh][key], 2 buffers
    __shared__ unsigned short Ps[4][16][64];    // per-wave P tile [q][key]

    const int t    = threadIdx.x;
    const int lane = t & 31;
    const int wave = t >> 5;
    const int lr   = lane & 15;
    const int lh   = lane >> 4;
    const int qt   = blockIdx.x;            // query tile
    const int bh   = blockIdx.y;            // b*8 + h
    const size_t headBase = (size_t)bh * 4096 * 64;   // elements

    const int waveId = __builtin_amdgcn_readfirstlane(wave);  // force scalar branch

    const unsigned long long Kga = (unsigned long long)(uintptr_t)(Kg + headBase);
    const unsigned long long Vga = (unsigned long long)(uintptr_t)(Vg + headBase);

    // Q A-frags (2 K-steps of 32 along Dh), kept in registers for whole loop
    Frag aq[2];
    {
        const unsigned short* Qrow =
            Q + headBase + (size_t)(qt * 64 + wave * 16 + lr) * 64;
        #pragma unroll
        for (int ks = 0; ks < 2; ++ks) {
            aq[ks].u[0] = *(const uint4*)&Qrow[ks * 32 + lh * 8];
            aq[ks].u[1] = *(const uint4*)&Qrow[ks * 32 + 16 + lh * 8];
        }
    }

    v8f acc[4] = {};
    float m[8], l[8];
    #pragma unroll
    for (int j = 0; j < 8; ++j) { m[j] = -1e30f; l[j] = 0.f; }

    // ---- TDM prologue: stage tile 0 into buffer 0 ----
    if (waveId == 0) {
        // K tile: 64 rows x 128B, fully contiguous -> flat 1024x1 copy (8B units)
        tdm_load_2d((unsigned)(uintptr_t)&Ks[0][0][0], Kga,
                    /*tile*/ 1024, 1, /*tensor*/ 65536, 1, /*stride*/ 65536);
        // V tile: 64 dh-rows x 128B, row stride = 4096*2B = 1024 units
        tdm_load_2d((unsigned)(uintptr_t)&Vs[0][0][0], Vga,
                    /*tile*/ 16, 64, /*tensor*/ 1024, 64, /*stride*/ 1024);
    }

    for (int kt = 0; kt < 64; ++kt) {
        const int buf = kt & 1;
        if (waveId == 0) __builtin_amdgcn_s_wait_tensorcnt(0);
        __syncthreads();   // publish tile kt; all waves done reading buf^1
        if (waveId == 0 && kt + 1 < 64) {
            // prefetch tile kt+1 into the other buffer during compute
            tdm_load_2d((unsigned)(uintptr_t)&Ks[buf ^ 1][0][0],
                        Kga + (unsigned long long)(kt + 1) * 8192ull,
                        1024, 1, 65536, 1, 65536);
            tdm_load_2d((unsigned)(uintptr_t)&Vs[buf ^ 1][0][0],
                        Vga + (unsigned long long)(kt + 1) * 128ull,
                        16, 64, 1024, 64, 1024);
        }

        // ---- scores: 16 queries x 64 keys, K-depth 64 (2 WMMA k-steps) ----
        v8f sc[4] = {};
        #pragma unroll
        for (int ks = 0; ks < 2; ++ks) {
            #pragma unroll
            for (int f = 0; f < 4; ++f) {
                Frag bk;  // col = key f*16+lr; K = dh ks*32 + {lh*8..+7, 16+lh*8..+7}
                bk.u[0] = *(const uint4*)&Ks[buf][f * 16 + lr][ks * 32 + lh * 8];
                bk.u[1] = *(const uint4*)&Ks[buf][f * 16 + lr][ks * 32 + 16 + lh * 8];
                sc[f] = __builtin_amdgcn_wmma_f32_16x16x32_bf16(
                    false, aq[ks].v, false, bk.v, (short)0, sc[f], false, false);
            }
        }

        // ---- online softmax (rows live in 16-lane halves, j=0..7) ----
        #pragma unroll
        for (int j = 0; j < 8; ++j) {
            float rmax = fmaxf(fmaxf(sc[0][j], sc[1][j]),
                               fmaxf(sc[2][j], sc[3][j]));
            #pragma unroll
            for (int off = 1; off < 16; off <<= 1)
                rmax = fmaxf(rmax, __shfl_xor(rmax, off, 32));
            float mn = fmaxf(m[j], rmax);
            float al = __expf(m[j] - mn);
            m[j] = mn;
            float rs = 0.f;
            #pragma unroll
            for (int f = 0; f < 4; ++f) {
                float p = __expf(sc[f][j] - mn);
                sc[f][j] = p;
                rs += p;
            }
            #pragma unroll
            for (int off = 1; off < 16; off <<= 1)
                rs += __shfl_xor(rs, off, 32);
            l[j] = l[j] * al + rs;
            #pragma unroll
            for (int f = 0; f < 4; ++f) acc[f][j] *= al;
        }

        // ---- P: C-layout -> bf16 LDS (per-wave tile; same-wave LDS in-order)
        #pragma unroll
        for (int f = 0; f < 4; ++f)
            #pragma unroll
            for (int j = 0; j < 8; ++j)
                Ps[wave][lh * 8 + j][f * 16 + lr] = f2bf(sc[f][j]);

        // ---- context += P @ V ----
        #pragma unroll
        for (int ks = 0; ks < 2; ++ks) {
            Frag ap;
            ap.u[0] = *(const uint4*)&Ps[wave][lr][ks * 32 + lh * 8];
            ap.u[1] = *(const uint4*)&Ps[wave][lr][ks * 32 + 16 + lh * 8];
            #pragma unroll
            for (int f = 0; f < 4; ++f) {
                Frag bv;  // col = dh f*16+lr; K = key ks*32 + {lh*8..+7, 16+lh*8..+7}
                bv.u[0] = *(const uint4*)&Vs[buf][f * 16 + lr][ks * 32 + lh * 8];
                bv.u[1] = *(const uint4*)&Vs[buf][f * 16 + lr][ks * 32 + 16 + lh * 8];
                acc[f] = __builtin_amdgcn_wmma_f32_16x16x32_bf16(
                    false, ap.v, false, bv.v, (short)0, acc[f], false, false);
            }
        }
    }

    // ---- epilogue: ctx[b, q, h*64 + d] = acc / l, stored bf16 ----
    const int b = bh >> 3, h = bh & 7;
    #pragma unroll
    for (int j = 0; j < 8; ++j) {
        float inv = 1.0f / l[j];
        int q = qt * 64 + wave * 16 + lh * 8 + j;
        size_t rowBase = ((size_t)b * 4096 + q) * 512 + h * 64;
        #pragma unroll
        for (int f = 0; f < 4; ++f)
            Ctx[rowBase + f * 16 + lr] = f2bf(acc[f][j] * inv);
    }
}

// ---------------------------------------------------------------------------
extern "C" void kernel_launch(void* const* d_in, const int* in_sizes, int n_in,
                              void* d_out, int out_size, void* d_ws, size_t ws_size,
                              hipStream_t stream)
{
    const float* X  = (const float*)d_in[0];
    const float* Wq = (const float*)d_in[1];
    const float* Wk = (const float*)d_in[2];
    const float* Wv = (const float*)d_in[3];
    const float* Wo = (const float*)d_in[4];

    unsigned short* ws = (unsigned short*)d_ws;
    const size_t NE = (size_t)8192 * 512;   // elements per [B,S,D] tensor
    unsigned short* Qw = ws;                // bf16 [B,H,S,Dh], pre-scaled by 1/8
    unsigned short* Kw = ws + NE;           // bf16 [B,H,S,Dh]
    unsigned short* Vw = ws + 2 * NE;       // bf16 [B,H,Dh,S]  (transposed)
    unsigned short* Cw = ws + 3 * NE;       // bf16 [B,S,D] context

    dim3 gg(8, 128);   // N/64, M/64
    dim3 bb(128);

    gemm512<false, 0><<<gg, bb, 0, stream>>>(X, Wq, Qw, 0.125f); // 1/sqrt(64)
    gemm512<false, 0><<<gg, bb, 0, stream>>>(X, Wk, Kw, 1.0f);
    gemm512<false, 2><<<gg, bb, 0, stream>>>(X, Wv, Vw, 1.0f);

    attn_kernel<<<dim3(64, 16), bb, 0, stream>>>(Qw, Kw, Vw, Cw);

    gemm512<true, 1><<<gg, bb, 0, stream>>>(Cw, Wo, d_out, 1.0f);
}
